// EdgeConv_56831007261032
// MI455X (gfx1250) — matrix operand order repaired and checked by reading, not compiled, not run on hardware
//
#include <hip/hip_runtime.h>

typedef _Float16 v8h  __attribute__((ext_vector_type(8)));
typedef _Float16 v16h __attribute__((ext_vector_type(16)));
typedef float    v8f  __attribute__((ext_vector_type(8)));
typedef float    v4f  __attribute__((ext_vector_type(4)));

#define BB   8
#define NNP  4096
#define DDIM 64
#define KNB  20
#define CCH  64
#define MPB  (NNP * KNB)      /* rows per batch: 81920 */
#define TPB  (MPB / 16)       /* 16-row tiles per batch: 5120 */
#define MTOT (BB * MPB)       /* 655360 */
#define CNTF ((float)(NNP * KNB))

// ---------- workspace layout (bytes) ----------
#define WS_W1H   ((size_t)0)          /* 64*128 f16 = 16384 */
#define WS_W2H   ((size_t)16384)      /* 64*64  f16 =  8192 */
#define WS_W3H   ((size_t)24576)      /* 64*64  f16 =  8192 */
#define WS_SUM   ((size_t)32768)      /* 512 f32 */
#define WS_SUMSQ ((size_t)34816)      /* 512 f32 */
#define WS_SCALE ((size_t)36864)      /* 512 f32 */
#define WS_BIAS  ((size_t)38912)      /* 512 f32 */
#define WS_HA    ((size_t)40960)      /* 655360*64 f16 = 83886080 */
#define WS_HB    (WS_HA + (size_t)83886080)

// ---------- small helpers ----------
static __device__ __forceinline__ v16h cat16(v8h lo, v8h hi) {
  return __builtin_shufflevector(lo, hi, 0,1,2,3,4,5,6,7,8,9,10,11,12,13,14,15);
}
static __device__ __forceinline__ v16h ldb32(const _Float16* p) {
  v8h lo = *(const v8h*)p;
  v8h hi = *(const v8h*)(p + 16);
  return cat16(lo, hi);
}
static __device__ __forceinline__ v8f ld8f(const float* p) {
  v4f a = *(const v4f*)p;
  v4f b = *(const v4f*)(p + 4);
  v8f r;
#pragma unroll
  for (int j = 0; j < 4; ++j) { r[j] = a[j]; r[j + 4] = b[j]; }
  return r;
}
static __device__ __forceinline__ v8h cvt8(const float* p) {
  v4f a = *(const v4f*)p;
  v4f b = *(const v4f*)(p + 4);
  v8h r;
#pragma unroll
  for (int j = 0; j < 4; ++j) { r[j] = (_Float16)a[j]; r[j + 4] = (_Float16)b[j]; }
  return r;
}
static __device__ __forceinline__ v8h cvt8_diff(const float* g, const float* c) {
  v4f a0 = *(const v4f*)g;        v4f a1 = *(const v4f*)(g + 4);
  v4f b0 = *(const v4f*)c;        v4f b1 = *(const v4f*)(c + 4);
  v8h r;
#pragma unroll
  for (int j = 0; j < 4; ++j) {
    r[j]     = (_Float16)(a0[j] - b0[j]);
    r[j + 4] = (_Float16)(a1[j] - b1[j]);
  }
  return r;
}
// normalize + leaky-relu a chunk of 8 input channels (f32 math, back to f16)
static __device__ __forceinline__ v8h norm8(v8h h, v8f s, v8f t) {
  v8h r;
#pragma unroll
  for (int j = 0; j < 8; ++j) {
    float f = (float)h[j] * s[j] + t[j];
    f = f > 0.f ? f : 0.01f * f;
    r[j] = (_Float16)f;
  }
  return r;
}
static __device__ __forceinline__ void atomAdd(float* p, float v) {
  __hip_atomic_fetch_add(p, v, __ATOMIC_RELAXED, __HIP_MEMORY_SCOPE_AGENT);
}

// ---------- utility kernels ----------
__global__ void cvt_f32_f16(const float* __restrict__ s, _Float16* __restrict__ d, int n) {
  int i = blockIdx.x * blockDim.x + threadIdx.x;
  if (i < n) d[i] = (_Float16)s[i];
}
__global__ void zero_f32(float* p, int n) {
  int i = blockIdx.x * blockDim.x + threadIdx.x;
  if (i < n) p[i] = 0.f;
}
__global__ void finalize_stats(const float* __restrict__ sum, const float* __restrict__ sumsq,
                               float* __restrict__ scale, float* __restrict__ bias) {
  int i = threadIdx.x;
  if (i < BB * CCH) {
    float mean = sum[i] / CNTF;
    float var  = sumsq[i] / CNTF - mean * mean;
    float rs   = rsqrtf(var + 1e-5f);
    scale[i] = rs;
    bias[i]  = -mean * rs;
  }
}

// ---------- stage 1: gather + feat + GEMM(128->64) + stats ----------
__global__ void __launch_bounds__(256)
stage1_kernel(const float* __restrict__ x, const long long* __restrict__ ind,
              const _Float16* __restrict__ w1h, _Float16* __restrict__ hout,
              float* __restrict__ sum, float* __restrict__ sumsq) {
  const int b    = blockIdx.y;
  const int wave = threadIdx.x >> 5;
  const int lane = threadIdx.x & 31;
  const int hi   = lane >> 4;
  const int ln   = lane & 15;
  const int klo  = hi ? 8 : 0;

  // B fragments: B(c,o) = W1[o,c]; lane ln holds column o=16*ot+ln, K in table order
  v16h bw[4][4];
#pragma unroll
  for (int kk = 0; kk < 4; ++kk)
#pragma unroll
    for (int ot = 0; ot < 4; ++ot)
      bw[kk][ot] = ldb32(w1h + (ot * 16 + ln) * 128 + kk * 32 + (hi ? 8 : 0));

  float s_acc[4] = {0.f, 0.f, 0.f, 0.f};
  float q_acc[4] = {0.f, 0.f, 0.f, 0.f};

  const int wpb = gridDim.x * (blockDim.x >> 5);
  const int wg  = blockIdx.x * (blockDim.x >> 5) + wave;

  for (int t = wg; t < TPB; t += wpb) {
    const int mBase = b * MPB + t * 16;
    const int m     = mBase + ln;                 // this lane's A row
    const long long gidx = ind[m];                // global gathered row
    const int crow = m / KNB;                     // center row (b*N + n)
    const float* gp = x + (long long)gidx * DDIM;
    const float* cp = x + (long long)crow * DDIM;

    v8f acc[4];
#pragma unroll
    for (int ot = 0; ot < 4; ++ot) acc[ot] = (v8f)(0.f);

#pragma unroll
    for (int kk = 0; kk < 4; ++kk) {
      const int p1 = kk * 32 + klo;
      const int p2 = p1 + 16;
      // feat[:, 0:64] = y - x_center ; feat[:, 64:128] = x_center
      v8h c0 = (p1 < 64) ? cvt8_diff(gp + p1, cp + p1) : cvt8(cp + (p1 - 64));
      v8h c1 = (p2 < 64) ? cvt8_diff(gp + p2, cp + p2) : cvt8(cp + (p2 - 64));
      v16h a = cat16(c0, c1);
#pragma unroll
      for (int ot = 0; ot < 4; ++ot)
        acc[ot] = __builtin_amdgcn_wmma_f32_16x16x32_f16(
            false, a, false, bw[kk][ot], (short)0, acc[ot], false, false);
    }

#pragma unroll
    for (int ot = 0; ot < 4; ++ot) {
      const int o = ln + 16 * ot;
#pragma unroll
      for (int rr = 0; rr < 8; ++rr) {
        const int row = mBase + rr + (hi ? 8 : 0);
        float v = acc[ot][rr];
        hout[(long long)row * CCH + o] = (_Float16)v;
        s_acc[ot] += v;
        q_acc[ot] += v * v;
      }
    }
  }
#pragma unroll
  for (int ot = 0; ot < 4; ++ot) {
    const int o = ln + 16 * ot;
    atomAdd(&sum[b * CCH + o], s_acc[ot]);
    atomAdd(&sumsq[b * CCH + o], q_acc[ot]);
  }
}

// ---------- stages 2/3: norm+lrelu of input, GEMM(64->64) + stats ----------
__global__ void __launch_bounds__(256)
stageN_kernel(const _Float16* __restrict__ hin, const _Float16* __restrict__ wh,
              const float* __restrict__ scale, const float* __restrict__ bias,
              _Float16* __restrict__ hout,
              float* __restrict__ sum, float* __restrict__ sumsq) {
  const int b    = blockIdx.y;
  const int wave = threadIdx.x >> 5;
  const int lane = threadIdx.x & 31;
  const int hi   = lane >> 4;
  const int ln   = lane & 15;
  const int klo  = hi ? 8 : 0;

  v16h bw[2][4];
#pragma unroll
  for (int kk = 0; kk < 2; ++kk)
#pragma unroll
    for (int ot = 0; ot < 4; ++ot)
      bw[kk][ot] = ldb32(wh + (ot * 16 + ln) * 64 + kk * 32 + (hi ? 8 : 0));

  // per-lane fixed input-channel chunks: klo, klo+16, klo+32, klo+48
  v8f sc[4], bi[4];
#pragma unroll
  for (int q = 0; q < 4; ++q) {
    sc[q] = ld8f(scale + b * CCH + klo + 16 * q);
    bi[q] = ld8f(bias  + b * CCH + klo + 16 * q);
  }

  float s_acc[4] = {0.f, 0.f, 0.f, 0.f};
  float q_acc[4] = {0.f, 0.f, 0.f, 0.f};

  const int wpb = gridDim.x * (blockDim.x >> 5);
  const int wg  = blockIdx.x * (blockDim.x >> 5) + wave;

  for (int t = wg; t < TPB; t += wpb) {
    const int mBase = b * MPB + t * 16;
    const _Float16* ap = hin + (long long)(mBase + ln) * CCH;

    v8f acc[4];
#pragma unroll
    for (int ot = 0; ot < 4; ++ot) acc[ot] = (v8f)(0.f);

#pragma unroll
    for (int kk = 0; kk < 2; ++kk) {
      v8h r0 = *(const v8h*)(ap + kk * 32 + klo);
      v8h r1 = *(const v8h*)(ap + kk * 32 + 16 + klo);
      r0 = norm8(r0, sc[2 * kk],     bi[2 * kk]);
      r1 = norm8(r1, sc[2 * kk + 1], bi[2 * kk + 1]);
      v16h a = cat16(r0, r1);
#pragma unroll
      for (int ot = 0; ot < 4; ++ot)
        acc[ot] = __builtin_amdgcn_wmma_f32_16x16x32_f16(
            false, a, false, bw[kk][ot], (short)0, acc[ot], false, false);
    }

#pragma unroll
    for (int ot = 0; ot < 4; ++ot) {
      const int o = ln + 16 * ot;
#pragma unroll
      for (int rr = 0; rr < 8; ++rr) {
        const int row = mBase + rr + (hi ? 8 : 0);
        float v = acc[ot][rr];
        hout[(long long)row * CCH + o] = (_Float16)v;
        s_acc[ot] += v;
        q_acc[ot] += v * v;
      }
    }
  }
#pragma unroll
  for (int ot = 0; ot < 4; ++ot) {
    const int o = ln + 16 * ot;
    atomAdd(&sum[b * CCH + o], s_acc[ot]);
    atomAdd(&sumsq[b * CCH + o], q_acc[ot]);
  }
}

// ---------- final: norm+lrelu of h3, max over k ----------
__global__ void maxpool_kernel(const _Float16* __restrict__ h3,
                               const float* __restrict__ scale, const float* __restrict__ bias,
                               float* __restrict__ out) {
  const int bn = blockIdx.x;              // 0 .. B*N-1
  const int o  = threadIdx.x;             // 0 .. 63
  const int b  = bn / NNP;
  const _Float16* p = h3 + (long long)bn * KNB * CCH + o;
  const float s = scale[b * CCH + o];
  const float t = bias[b * CCH + o];
  float m = -3.402823466e38f;
#pragma unroll 4
  for (int k = 0; k < KNB; ++k) {
    float f = (float)p[(long long)k * CCH] * s + t;
    f = f > 0.f ? f : 0.01f * f;
    m = fmaxf(m, f);
  }
  out[(long long)bn * CCH + o] = m;
}

extern "C" void kernel_launch(void* const* d_in, const int* in_sizes, int n_in,
                              void* d_out, int out_size, void* d_ws, size_t ws_size,
                              hipStream_t stream) {
  (void)in_sizes; (void)n_in; (void)out_size; (void)ws_size;
  const float*     x   = (const float*)d_in[0];
  const long long* ind = (const long long*)d_in[1];
  const float*     W1  = (const float*)d_in[2];
  const float*     W2  = (const float*)d_in[3];
  const float*     W3  = (const float*)d_in[4];
  float* out = (float*)d_out;

  char* ws = (char*)d_ws;
  _Float16* w1h   = (_Float16*)(ws + WS_W1H);
  _Float16* w2h   = (_Float16*)(ws + WS_W2H);
  _Float16* w3h   = (_Float16*)(ws + WS_W3H);
  float*    sum   = (float*)(ws + WS_SUM);
  float*    sumsq = (float*)(ws + WS_SUMSQ);
  float*    scale = (float*)(ws + WS_SCALE);
  float*    bias  = (float*)(ws + WS_BIAS);
  _Float16* hA    = (_Float16*)(ws + WS_HA);
  _Float16* hB    = (_Float16*)(ws + WS_HB);

  // convert weights to f16
  cvt_f32_f16<<<32, 256, 0, stream>>>(W1, w1h, 64 * 128);
  cvt_f32_f16<<<16, 256, 0, stream>>>(W2, w2h, 64 * 64);
  cvt_f32_f16<<<16, 256, 0, stream>>>(W3, w3h, 64 * 64);

  dim3 grid(160, BB);   // blockIdx.y = batch (keeps stats accumulation local)
  const int blk = 256;  // 8 waves

  // stage 1
  zero_f32<<<1, 1024, 0, stream>>>(sum, 1024);  // sum+sumsq contiguous
  stage1_kernel<<<grid, blk, 0, stream>>>(x, ind, w1h, hA, sum, sumsq);
  finalize_stats<<<1, 512, 0, stream>>>(sum, sumsq, scale, bias);

  // stage 2
  zero_f32<<<1, 1024, 0, stream>>>(sum, 1024);
  stageN_kernel<<<grid, blk, 0, stream>>>(hA, w2h, scale, bias, hB, sum, sumsq);
  finalize_stats<<<1, 512, 0, stream>>>(sum, sumsq, scale, bias);

  // stage 3
  zero_f32<<<1, 1024, 0, stream>>>(sum, 1024);
  stageN_kernel<<<grid, blk, 0, stream>>>(hB, w3h, scale, bias, hA, sum, sumsq);
  finalize_stats<<<1, 512, 0, stream>>>(sum, sumsq, scale, bias);

  // final norm + lrelu + max over k
  maxpool_kernel<<<BB * NNP, 64, 0, stream>>>(hA, scale, bias, out);
}
